// SlotAttentionFG_10041633538397
// MI455X (gfx1250) — matrix operand order repaired and checked by reading, not compiled, not run on hardware
//
#include <hip/hip_runtime.h>
#include <hip/hip_bf16.h>

#define B_   16
#define H_   64
#define W2_  64
#define D_   64
#define C_   16
#define K_   5
#define N_   4096
#define ITERS_ 4
#define CH_  256                        // k rows staged per async chunk
#define NCH_ (N_ / CH_)                 // 16 chunks
#define OUT_FG   (B_*K_*(D_+C_))
#define OUT_ATTN (OUT_FG + B_*K_*2)

typedef __attribute__((ext_vector_type(16))) __bf16 v16bf;
typedef __attribute__((ext_vector_type(8)))  float  v8f;

#define DEV static __device__ __forceinline__

DEV __bf16 f2bf(float f) {
  unsigned u = __builtin_bit_cast(unsigned, f);
  u += 0x7fffu + ((u >> 16) & 1u);           // round-to-nearest-even
  unsigned short h = (unsigned short)(u >> 16);
  return __builtin_bit_cast(__bf16, h);
}

// A-matrix tile 16x32 bf16: lane L (g=L>>4, m=L&15), vgpr p<4 -> K=g*8+2p, p>=4 -> K=16+g*8+2(p-4)
DEV v16bf packA(const float* src, int lda, int lane) {
  int g = (lane >> 4) & 1, m = lane & 15;
  v16bf a;
#pragma unroll
  for (int p = 0; p < 8; ++p) {
    int kk = (p < 4) ? (g * 8 + 2 * p) : (16 + g * 8 + 2 * (p - 4));
    const float* s = src + m * lda + kk;
    a[2 * p]     = f2bf(s[0]);
    a[2 * p + 1] = f2bf(s[1]);
  }
  return a;
}

// B-matrix tile 32x16 bf16: lane L (g=L>>4, n=L&15); element j<8 -> K = g*8+j, j>=8 -> K = 16+g*8+(j-8), col n.
// B[k][n] = W[n][k] for X @ W^T with W row-major (out,in).
DEV v16bf packB(const float* w, int ldw, int lane) {
  int g = (lane >> 4) & 1, n = lane & 15;
  v16bf bt;
#pragma unroll
  for (int j = 0; j < 16; ++j) bt[j] = f2bf(w[n * ldw + ((j < 8) ? (g * 8 + j) : (16 + g * 8 + (j - 8)))]);
  return bt;
}

// C/D 16x16 f32: lane L (g=L>>4, n=L&15), vgpr v -> row m = g*8+v
DEV void storeD(float* dst, int ldd, int lane, v8f acc) {
  int g = (lane >> 4) & 1, n = lane & 15;
#pragma unroll
  for (int v = 0; v < 8; ++v) dst[(g * 8 + v) * ldd + n] = acc[v];
}

DEV v8f wmma_bf16(v16bf a, v16bf b, v8f c) {
  return __builtin_amdgcn_wmma_f32_16x16x32_bf16(false, a, false, b, (short)0, c, false, false);
}

// Async bulk copy: 64KB (CH_ rows x 256B) global -> LDS via GLOBAL_LOAD_ASYNC_TO_LDS_B128.
// 256 threads x 16 issues x 16B. Tracked on ASYNCcnt (16 per wave per chunk).
DEV void async_chunk(const float* gsrc, const float* ldst, int tid) {
  unsigned lbase = (unsigned)(uintptr_t)ldst;
  unsigned long long gbase = (unsigned long long)(uintptr_t)gsrc;
#pragma unroll
  for (int i = 0; i < 16; ++i) {
    unsigned e = (unsigned)(i * 256 + tid);
    unsigned loff = lbase + e * 16u;
    unsigned long long ga = gbase + (unsigned long long)e * 16ull;
    asm volatile("global_load_async_to_lds_b128 %0, %1, off"
                 :: "v"(loff), "v"(ga) : "memory");
  }
}

// ---------------- small prep kernels ----------------

__global__ void fgpos_kernel(const float* __restrict__ mask, float* __restrict__ fg) {
  int ks = blockIdx.x, lane = threadIdx.x;
  float ms = 0.f, sx = 0.f, sy = 0.f;
  for (int n = lane; n < N_; n += 32) {
    int hi = n >> 6, wi = n & 63;
    float gx = -1.f + 2.f * wi / 63.f;
    float gy = -1.f + 2.f * hi / 63.f;
    float m = mask[ks * N_ + n];
    ms += m; sx += m * gx; sy += m * gy;
  }
  for (int o = 16; o; o >>= 1) {
    ms += __shfl_xor(ms, o); sx += __shfl_xor(sx, o); sy += __shfl_xor(sy, o);
  }
  if (lane == 0) {
    float inv = 1.f / (ms + 1e-5f);
    fg[ks * 2 + 0] = sx * inv;
    fg[ks * 2 + 1] = sy * inv;
  }
}

__global__ void fgout_kernel(const float* __restrict__ fg, float* __restrict__ out) {
  int t = threadIdx.x;
  if (t >= B_ * K_ * 2) return;
  int j = t & 1, ks = (t >> 1) % K_;
  out[OUT_FG + t] = fg[ks * 2 + j];
}

__global__ void ln64_kernel(const float* __restrict__ x, const float* __restrict__ g,
                            const float* __restrict__ b, float* __restrict__ y) {
  int row = blockIdx.x, lane = threadIdx.x;
  const float* xr = x + (size_t)row * 64;
  float x0 = xr[lane], x1 = xr[lane + 32];
  float s = x0 + x1, s2 = x0 * x0 + x1 * x1;
  for (int o = 16; o; o >>= 1) { s += __shfl_xor(s, o); s2 += __shfl_xor(s2, o); }
  float m = s * (1.f / 64.f);
  float rstd = rsqrtf(s2 * (1.f / 64.f) - m * m + 1e-5f);
  float* yr = y + (size_t)row * 64;
  yr[lane]      = (x0 - m) * rstd * g[lane]      + b[lane];
  yr[lane + 32] = (x1 - m) * rstd * g[lane + 32] + b[lane + 32];
}

__global__ void ln16_kernel(const float* __restrict__ x, const float* __restrict__ g,
                            const float* __restrict__ b, float* __restrict__ y) {
  int lane = threadIdx.x;
  int row = blockIdx.x * 2 + (lane >> 4);
  int c = lane & 15;
  float v = x[(size_t)row * 16 + c];
  float s = v, s2 = v * v;
  for (int o = 8; o; o >>= 1) { s += __shfl_xor(s, o, 16); s2 += __shfl_xor(s2, o, 16); }
  float m = s * (1.f / 16.f);
  float rstd = rsqrtf(s2 * (1.f / 16.f) - m * m + 1e-5f);
  y[(size_t)row * 16 + c] = (v - m) * rstd * g[c] + b[c];
}

__global__ void slotinit_kernel(const float* __restrict__ noise, const float* __restrict__ mu,
                                const float* __restrict__ ls, float* __restrict__ slot) {
  int i = blockIdx.x * blockDim.x + threadIdx.x;
  if (i >= B_ * K_ * D_) return;
  int d = i & 63;
  slot[i] = mu[d] + expf(ls[d]) * noise[i];
}

// ---------------- WMMA GEMM kernels ----------------

// C[M x Nout] = A[M x Kin] @ W^T (+bias). grid (M/16, Nout/16), block 32 (one wave).
__global__ void gemm_bias_kernel(const float* __restrict__ A, const float* __restrict__ W,
                                 const float* __restrict__ bias, float* __restrict__ C,
                                 int Nout, int Kin) {
  int lane = threadIdx.x;
  int m0 = blockIdx.x * 16, n0 = blockIdx.y * 16;
  float bv = bias ? bias[n0 + (lane & 15)] : 0.f;
  v8f acc;
#pragma unroll
  for (int v = 0; v < 8; ++v) acc[v] = bv;
  for (int k0 = 0; k0 < Kin; k0 += 32) {
    v16bf a = packA(A + (size_t)m0 * Kin + k0, Kin, lane);
    v16bf w = packB(W + (size_t)n0 * Kin + k0, Kin, lane);
    acc = wmma_bf16(a, w, acc);
  }
  storeD(C + (size_t)m0 * Nout + n0, Nout, lane, acc);
}

// k/v: LN(featX + grid_embed) @ w_mlp^T + b_mlp. grid (N/16, K, B), block 32.
__global__ void kv_kernel(const float* __restrict__ featk, const float* __restrict__ featv,
                          const float* __restrict__ fg,
                          const float* __restrict__ w_grid, const float* __restrict__ b_grid,
                          const float* __restrict__ lng, const float* __restrict__ lnb,
                          const float* __restrict__ w_mlp, const float* __restrict__ b_mlp,
                          float* __restrict__ kmat, float* __restrict__ vmat) {
  __shared__ float st[1024];
  __shared__ float mu[16], rs[16];
  int lane = threadIdx.x;
  int n0 = blockIdx.x * 16, ks = blockIdx.y, b = blockIdx.z;
  float fgx = fg[ks * 2 + 0], fgy = fg[ks * 2 + 1];
  size_t srcbase = ((size_t)b * N_ + n0) * 64;
  size_t dstbase = (((size_t)(b * K_ + ks)) * N_ + n0) * 64;
  for (int pass = 0; pass < 2; ++pass) {
    const float* src = (pass == 0 ? featk : featv) + srcbase;
    float* dst = (pass == 0 ? kmat : vmat) + dstbase;
    for (int idx = lane; idx < 1024; idx += 32) {
      int r = idx >> 6, d = idx & 63;
      int n = n0 + r;
      int hi = n >> 6, wi = n & 63;
      float rx = (-1.f + 2.f * wi / 63.f) - fgx;
      float ry = (-1.f + 2.f * hi / 63.f) - fgy;
      float ge = rx * (w_grid[d * 4 + 0] - w_grid[d * 4 + 2]) +
                 ry * (w_grid[d * 4 + 1] - w_grid[d * 4 + 3]) + b_grid[d];
      st[idx] = src[idx] + ge;
    }
    __syncthreads();
    if (lane < 16) {
      float s = 0.f, s2 = 0.f;
      for (int d = 0; d < 64; ++d) { float xx = st[lane * 64 + d]; s += xx; s2 += xx * xx; }
      float m = s * (1.f / 64.f);
      mu[lane] = m;
      rs[lane] = rsqrtf(s2 * (1.f / 64.f) - m * m + 1e-5f);
    }
    __syncthreads();
    for (int idx = lane; idx < 1024; idx += 32) {
      int r = idx >> 6, d = idx & 63;
      st[idx] = (st[idx] - mu[r]) * rs[r] * lng[d] + lnb[d];
    }
    __syncthreads();
    for (int n4 = 0; n4 < 4; ++n4) {
      float bv = b_mlp[n4 * 16 + (lane & 15)];
      v8f acc;
#pragma unroll
      for (int v = 0; v < 8; ++v) acc[v] = bv;
#pragma unroll
      for (int k0 = 0; k0 < 64; k0 += 32) {
        v16bf a = packA(st + k0, 64, lane);
        v16bf w = packB(w_mlp + n4 * 16 * 64 + k0, 64, lane);
        acc = wmma_bf16(a, w, acc);
      }
      storeD(dst + n4 * 16, 64, lane, acc);
    }
    __syncthreads();
  }
}

// out[16 rows x 64] = LN(X rows) @ W^T (+bias) (+res). grid (5), block 32.
__global__ void rowln_gemm_kernel(const float* __restrict__ X, const float* __restrict__ lng,
                                  const float* __restrict__ lnb, const float* __restrict__ W,
                                  const float* __restrict__ bias, const float* __restrict__ res,
                                  float* __restrict__ out) {
  __shared__ float st[1024];
  __shared__ float mu[16], rs[16];
  int lane = threadIdx.x;
  int m0 = blockIdx.x * 16;
  for (int idx = lane; idx < 1024; idx += 32) st[idx] = X[(size_t)m0 * 64 + idx];
  __syncthreads();
  if (lane < 16) {
    float s = 0.f, s2 = 0.f;
    for (int d = 0; d < 64; ++d) { float xx = st[lane * 64 + d]; s += xx; s2 += xx * xx; }
    float m = s * (1.f / 64.f);
    mu[lane] = m;
    rs[lane] = rsqrtf(s2 * (1.f / 64.f) - m * m + 1e-5f);
  }
  __syncthreads();
  for (int idx = lane; idx < 1024; idx += 32) {
    int r = idx >> 6, d = idx & 63;
    st[idx] = (st[idx] - mu[r]) * rs[r] * lng[d] + lnb[d];
  }
  __syncthreads();
  for (int n4 = 0; n4 < 4; ++n4) {
    float bv = bias ? bias[n4 * 16 + (lane & 15)] : 0.f;
    v8f acc;
#pragma unroll
    for (int v = 0; v < 8; ++v) acc[v] = bv;
#pragma unroll
    for (int k0 = 0; k0 < 64; k0 += 32) {
      v16bf a = packA(st + k0, 64, lane);
      v16bf w = packB(W + n4 * 16 * 64 + k0, 64, lane);
      acc = wmma_bf16(a, w, acc);
    }
    int gg = (lane >> 4) & 1, n = lane & 15;
#pragma unroll
    for (int v = 0; v < 8; ++v) {
      int gi_ = (m0 + gg * 8 + v) * 64 + n4 * 16 + n;
      float val = acc[v];
      if (res) val += res[gi_];
      out[gi_] = val;
    }
  }
}

// ---------------- attention kernels ----------------

// logits + softmax. k is double-buffered through LDS with GLOBAL_LOAD_ASYNC_TO_LDS_B128,
// DMA of chunk c+1 overlapped with dot products on chunk c. grid (K,B), block 256.
__global__ void attn_kernel(const float* __restrict__ q, const float* __restrict__ kmat,
                            float* __restrict__ attn) {
  __shared__ float qs[64];
  __shared__ float red[256];
  __shared__ float kbuf[2][CH_ * 64];   // 2 x 64KB
  int ks = blockIdx.x, b = blockIdx.y, tid = threadIdx.x;
  int bk = b * K_ + ks;
  if (tid < 64) qs[tid] = q[bk * 64 + tid];
  const float* kb = kmat + (size_t)bk * N_ * 64;
  float logit[NCH_];
  async_chunk(kb, kbuf[0], tid);                       // prologue DMA
  for (int c = 0; c < NCH_; ++c) {
    if (c + 1 < NCH_) {
      async_chunk(kb + (size_t)(c + 1) * CH_ * 64, kbuf[(c + 1) & 1], tid);
      asm volatile("s_wait_asynccnt 16" ::: "memory"); // chunk c's 16 loads done (in-order)
    } else {
      asm volatile("s_wait_asynccnt 0" ::: "memory");
    }
    __syncthreads();                                   // all waves' DMA portions visible
    const float* kr = kbuf[c & 1] + tid * 64;
    int rot = (tid & 31) * 2;                          // bank-conflict-free rotation
    float s = 0.f;
#pragma unroll
    for (int j = 0; j < 64; ++j) { int d = (rot + j) & 63; s += qs[d] * kr[d]; }
    logit[c] = s * 0.125f;                             // D^-0.5
    __syncthreads();                                   // buffer safe to overwrite
  }
  float lmax = -3.0e38f;
#pragma unroll
  for (int c = 0; c < NCH_; ++c) lmax = fmaxf(lmax, logit[c]);
  red[tid] = lmax; __syncthreads();
  for (int o = 128; o; o >>= 1) { if (tid < o) red[tid] = fmaxf(red[tid], red[tid + o]); __syncthreads(); }
  float m = red[0]; __syncthreads();
  float ls = 0.f;
#pragma unroll
  for (int c = 0; c < NCH_; ++c) { logit[c] = expf(logit[c] - m); ls += logit[c]; }
  red[tid] = ls; __syncthreads();
  for (int o = 128; o; o >>= 1) { if (tid < o) red[tid] += red[tid + o]; __syncthreads(); }
  float inv = 1.f / red[0];
  float* ab = attn + (size_t)bk * N_;
#pragma unroll
  for (int c = 0; c < NCH_; ++c) ab[c * CH_ + tid] = logit[c] * inv;
}

__global__ void upd_kernel(const float* __restrict__ attn, const float* __restrict__ vmat,
                           float* __restrict__ upd) {
  __shared__ float red[256];
  int ks = blockIdx.x, b = blockIdx.y, tid = threadIdx.x;
  int d = tid & 63, g = tid >> 6;
  int bk = b * K_ + ks;
  const float* ab = attn + (size_t)bk * N_;
  const float* vb = vmat + (size_t)bk * N_ * 64;
  float s = 0.f;
  for (int n = g; n < N_; n += 4) s += ab[n] * vb[(size_t)n * 64 + d];
  red[tid] = s; __syncthreads();
  if (tid < 64) upd[bk * 64 + tid] = red[tid] + red[tid + 64] + red[tid + 128] + red[tid + 192];
}

__global__ void color_kernel(const float* __restrict__ attn, const float* __restrict__ featc,
                             float* __restrict__ out) {
  __shared__ float red[256];
  int ks = blockIdx.x, b = blockIdx.y, tid = threadIdx.x;
  int c = tid & 15, g = tid >> 4;
  int bk = b * K_ + ks;
  const float* ab = attn + (size_t)bk * N_;
  const float* fc = featc + (size_t)b * N_ * 16;
  float s = 0.f;
  for (int n = g; n < N_; n += 16) s += ab[n] * fc[(size_t)n * 16 + c];
  red[tid] = s; __syncthreads();
  for (int o = 128; o >= 16; o >>= 1) { if (tid < o) red[tid] += red[tid + o]; __syncthreads(); }
  if (tid < 16) out[bk * (D_ + C_) + D_ + tid] = red[tid];
}

__global__ void gru_kernel(const float* __restrict__ gi, const float* __restrict__ gh,
                           const float* __restrict__ slot, float* __restrict__ ns) {
  int i = blockIdx.x * blockDim.x + threadIdx.x;
  if (i >= B_ * K_ * D_) return;
  int row = i >> 6, d = i & 63;
  float ir = gi[row * 192 + d], iz = gi[row * 192 + 64 + d], in_ = gi[row * 192 + 128 + d];
  float hr = gh[row * 192 + d], hz = gh[row * 192 + 64 + d], hn = gh[row * 192 + 128 + d];
  float r = 1.f / (1.f + expf(-(ir + hr)));
  float z = 1.f / (1.f + expf(-(iz + hz)));
  float nn = tanhf(in_ + r * hn);
  float hp = slot[i];
  ns[i] = (1.f - z) * nn + z * hp;
}

__global__ void slotout_kernel(const float* __restrict__ slot, float* __restrict__ out) {
  int i = blockIdx.x * blockDim.x + threadIdx.x;
  if (i >= B_ * K_ * D_) return;
  int row = i >> 6, d = i & 63;
  out[row * (D_ + C_) + d] = slot[i];
}

__global__ void attn_norm_kernel(const float* __restrict__ attn, float* __restrict__ out) {
  __shared__ float rmin[256], rmax[256];
  int ks = blockIdx.x, b = blockIdx.y, tid = threadIdx.x;
  int bk = b * K_ + ks;
  const float* a = attn + (size_t)bk * N_;
  float mn = 3.0e38f, mx = -3.0e38f;
#pragma unroll
  for (int i = 0; i < 16; ++i) {
    float v = a[tid + i * 256];
    mn = fminf(mn, v); mx = fmaxf(mx, v);
  }
  rmin[tid] = mn; rmax[tid] = mx; __syncthreads();
  for (int o = 128; o; o >>= 1) {
    if (tid < o) { rmin[tid] = fminf(rmin[tid], rmin[tid + o]); rmax[tid] = fmaxf(rmax[tid], rmax[tid + o]); }
    __syncthreads();
  }
  float lo = rmin[0];
  float inv = 1.f / (rmax[0] - lo + 1e-5f);
  float* o = out + OUT_ATTN + (size_t)bk * N_;
#pragma unroll
  for (int i = 0; i < 16; ++i) { int n = tid + i * 256; o[n] = (a[n] - lo) * inv; }
}

// ---------------- host orchestration ----------------

extern "C" void kernel_launch(void* const* d_in, const int* in_sizes, int n_in,
                              void* d_out, int out_size, void* d_ws, size_t ws_size,
                              hipStream_t stream) {
  (void)in_sizes; (void)n_in; (void)out_size;
  const float* feat       = (const float*)d_in[0];
  const float* feat_color = (const float*)d_in[1];
  const float* mask       = (const float*)d_in[2];
  const float* slot_noise = (const float*)d_in[3];
  const float* slots_mu   = (const float*)d_in[4];
  const float* slots_ls   = (const float*)d_in[5];
  const float* w_grid     = (const float*)d_in[6];
  const float* b_grid     = (const float*)d_in[7];
  const float* w_k        = (const float*)d_in[8];
  const float* w_v        = (const float*)d_in[9];
  const float* ln_kv_g    = (const float*)d_in[10];
  const float* ln_kv_b    = (const float*)d_in[11];
  const float* w_mlp      = (const float*)d_in[12];
  const float* b_mlp      = (const float*)d_in[13];
  const float* ln_q_g     = (const float*)d_in[14];
  const float* ln_q_b     = (const float*)d_in[15];
  const float* w_q        = (const float*)d_in[16];
  const float* w_ih       = (const float*)d_in[17];
  const float* w_hh       = (const float*)d_in[18];
  const float* b_ih       = (const float*)d_in[19];
  const float* b_hh       = (const float*)d_in[20];
  const float* ln_feat_g  = (const float*)d_in[21];
  const float* ln_feat_b  = (const float*)d_in[22];
  const float* ln_color_g = (const float*)d_in[23];
  const float* ln_color_b = (const float*)d_in[24];
  const float* ln_res_g   = (const float*)d_in[25];
  const float* ln_res_b   = (const float*)d_in[26];
  const float* w_res      = (const float*)d_in[27];
  const float* b_res      = (const float*)d_in[28];

  float* ws  = (float*)d_ws;
  float* out = (float*)d_out;

  const size_t oFG    = 0;
  const size_t oFEATN = 16;
  const size_t oFEATC = oFEATN + (size_t)B_ * N_ * 64;
  const size_t oFEATK = oFEATC + (size_t)B_ * N_ * 16;
  const size_t oFEATV = oFEATK + (size_t)B_ * N_ * 64;
  const size_t oKMAT  = oFEATV + (size_t)B_ * N_ * 64;
  const size_t oVMAT  = oKMAT + (size_t)B_ * K_ * N_ * 64;
  const size_t oSLOT  = oVMAT + (size_t)B_ * K_ * N_ * 64;
  const size_t oQ     = oSLOT + B_ * K_ * 64;
  const size_t oNS    = oQ    + B_ * K_ * 64;
  const size_t oGI    = oNS   + B_ * K_ * 64;
  const size_t oGH    = oGI   + B_ * K_ * 192;
  const size_t oUPD   = oGH   + B_ * K_ * 192;
  const size_t oATTN  = oUPD  + B_ * K_ * 64;
  const size_t total  = oATTN + (size_t)B_ * K_ * N_;
  if (ws_size < total * sizeof(float)) return;

  float* FG    = ws + oFG;
  float* FEATN = ws + oFEATN;
  float* FEATC = ws + oFEATC;
  float* FEATK = ws + oFEATK;
  float* FEATV = ws + oFEATV;
  float* KMAT  = ws + oKMAT;
  float* VMAT  = ws + oVMAT;
  float* SLOT  = ws + oSLOT;
  float* Q     = ws + oQ;
  float* NS    = ws + oNS;
  float* GI    = ws + oGI;
  float* GH    = ws + oGH;
  float* UPD   = ws + oUPD;
  float* ATTN  = ws + oATTN;

  fgpos_kernel<<<K_, 32, 0, stream>>>(mask, FG);
  fgout_kernel<<<1, 160, 0, stream>>>(FG, out);
  ln64_kernel<<<B_ * N_, 32, 0, stream>>>(feat, ln_feat_g, ln_feat_b, FEATN);
  ln16_kernel<<<B_ * N_ / 2, 32, 0, stream>>>(feat_color, ln_color_g, ln_color_b, FEATC);
  slotinit_kernel<<<(B_ * K_ * D_ + 255) / 256, 256, 0, stream>>>(slot_noise, slots_mu, slots_ls, SLOT);

  // featk = feat_n @ w_k^T ; featv = feat_n @ w_v^T   (WMMA)
  gemm_bias_kernel<<<dim3(B_ * N_ / 16, 4), 32, 0, stream>>>(FEATN, w_k, nullptr, FEATK, 64, 64);
  gemm_bias_kernel<<<dim3(B_ * N_ / 16, 4), 32, 0, stream>>>(FEATN, w_v, nullptr, FEATV, 64, 64);

  // k, v = LN(featX + grid_embed) @ w_mlp^T + b_mlp    (WMMA, fused)
  kv_kernel<<<dim3(N_ / 16, K_, B_), 32, 0, stream>>>(FEATK, FEATV, FG, w_grid, b_grid,
                                                      ln_kv_g, ln_kv_b, w_mlp, b_mlp, KMAT, VMAT);

  for (int it = 0; it < ITERS_; ++it) {
    // q = LN(slot) @ w_q^T   (WMMA)
    rowln_gemm_kernel<<<B_ * K_ / 16, 32, 0, stream>>>(SLOT, ln_q_g, ln_q_b, w_q,
                                                       nullptr, nullptr, Q);
    attn_kernel<<<dim3(K_, B_), 256, 0, stream>>>(Q, KMAT, ATTN);
    if (it < ITERS_ - 1) {
      upd_kernel<<<dim3(K_, B_), 256, 0, stream>>>(ATTN, VMAT, UPD);
      // GRU gates (WMMA): gi = upd @ w_ih^T + b_ih ; gh = slot @ w_hh^T + b_hh
      gemm_bias_kernel<<<dim3(B_ * K_ / 16, 12), 32, 0, stream>>>(UPD, w_ih, b_ih, GI, 192, 64);
      gemm_bias_kernel<<<dim3(B_ * K_ / 16, 12), 32, 0, stream>>>(SLOT, w_hh, b_hh, GH, 192, 64);
      gru_kernel<<<(B_ * K_ * D_ + 255) / 256, 256, 0, stream>>>(GI, GH, SLOT, NS);
      // slot = LN(newslot) @ w_res^T + b_res + prev   (WMMA)
      rowln_gemm_kernel<<<B_ * K_ / 16, 32, 0, stream>>>(NS, ln_res_g, ln_res_b, w_res,
                                                         b_res, SLOT, SLOT);
    } else {
      color_kernel<<<dim3(K_, B_), 256, 0, stream>>>(ATTN, FEATC, out);
      slotout_kernel<<<(B_ * K_ * D_ + 255) / 256, 256, 0, stream>>>(SLOT, out);
      attn_norm_kernel<<<dim3(K_, B_), 256, 0, stream>>>(ATTN, out);
    }
  }
}
